// GunnarODE_14370960572829
// MI455X (gfx1250) — matrix operand order repaired and checked by reading, not compiled, not run on hardware
//
#include <hip/hip_runtime.h>

#define LSEQ 512
#define BB   4096
#define UU   8
#define YY   16
#define HH   128
#define CCH  9      // U+1
#define FDIM 144    // Y*C

typedef _Float16 v16h __attribute__((ext_vector_type(16)));
typedef float    v8f  __attribute__((ext_vector_type(8)));

// dword offsets inside ws / LDS image
#define W1A_OFF 0            //  8 tiles * 32 lanes * 8 dwords = 2048
#define W2A_OFF 2048         // 36 tiles * 32 lanes * 8 dwords = 9216
#define B2C_OFF 11264        //  9 tiles * 32 lanes * 8 dwords = 2304
#define LDS_DWORDS 13568     // total (54,272 bytes)

__device__ __forceinline__ float swap16(float x) {
  // swap lane halves 0-15 <-> 16-31 (xor 0x10); group-of-32 ds_swizzle pattern
  int i = __builtin_bit_cast(int, x);
  i = __builtin_amdgcn_ds_swizzle(i, 0x401F);
  return __builtin_bit_cast(float, i);
}

__device__ __forceinline__ float tanh_fast(float x) {
#if __has_builtin(__builtin_amdgcn_tanhf)
  return __builtin_amdgcn_tanhf(x);
#else
  float e = __builtin_amdgcn_exp2f(x * 2.885390081777927f); // exp(2x)
  return (e - 1.0f) * __builtin_amdgcn_rcpf(e + 1.0f);
#endif
}

// ---------------------------------------------------------------------------
// Prep kernel: pack W1/b1/W2/b2 into exact WMMA per-lane operand layouts.
//  W1A[i][lane][v] : f16 A-operand tiles (16x32), K=0..15 = W1 row, K=16 = b1
//  W2A[j*4+c][lane][v] : f16 A-operand tiles of W2 (K chunk c)
//  B2C[j][lane][v] : f32 C-operand bias tiles (b2) in C/D layout
// A-layout as packed (16-bit, 16x32): lanes0-15 V0-7:K0..15;
//                           lanes16-31 V0-7:K16..31
// ---------------------------------------------------------------------------
__global__ void prep_weights(const float* __restrict__ W1, const float* __restrict__ b1,
                             const float* __restrict__ W2, const float* __restrict__ b2,
                             unsigned int* __restrict__ ws) {
  int idx = blockIdx.x * blockDim.x + threadIdx.x;
  if (idx >= LDS_DWORDS) return;
  if (idx < W2A_OFF) {                          // ---- W1A ----
    int i    = idx >> 8;
    int lane = (idx >> 3) & 31;
    int v    = idx & 7;
    int hi   = (lane >= 16) ? 1 : 0;
    int m    = 16 * i + (lane & 15);            // H feature row
    int k0   = 16 * hi + 2 * v;
    int k1   = k0 + 1;
    float f0 = (k0 < YY) ? W1[m * YY + k0] : ((k0 == YY) ? b1[m] : 0.0f);
    float f1 = (k1 < YY) ? W1[m * YY + k1] : ((k1 == YY) ? b1[m] : 0.0f);
    unsigned short h0 = __builtin_bit_cast(unsigned short, (_Float16)f0);
    unsigned short h1 = __builtin_bit_cast(unsigned short, (_Float16)f1);
    ws[idx] = (unsigned int)h0 | ((unsigned int)h1 << 16);
  } else if (idx < B2C_OFF) {                   // ---- W2A ----
    int t    = idx - W2A_OFF;
    int jt   = t >> 8;                          // j*4 + c, 0..35
    int lane = (t >> 3) & 31;
    int v    = t & 7;
    int hi   = (lane >= 16) ? 1 : 0;
    int j    = jt >> 2, c = jt & 3;
    int m    = 16 * j + (lane & 15);            // output feature row (0..143)
    int k0   = 32 * c + 16 * hi + 2 * v;
    float f0 = W2[m * HH + k0];
    float f1 = W2[m * HH + k0 + 1];
    unsigned short h0 = __builtin_bit_cast(unsigned short, (_Float16)f0);
    unsigned short h1 = __builtin_bit_cast(unsigned short, (_Float16)f1);
    ws[idx] = (unsigned int)h0 | ((unsigned int)h1 << 16);
  } else {                                      // ---- B2C (b2 in C/D layout) ----
    int t    = idx - B2C_OFF;
    int j    = t >> 8;
    int lane = (t >> 3) & 31;
    int v    = t & 7;
    int hi   = (lane >= 16) ? 1 : 0;
    int m    = 16 * j + v + 8 * hi;
    ws[idx]  = __builtin_bit_cast(unsigned int, b2[m]);
  }
}

// ---------------------------------------------------------------------------
// Main kernel: each wave owns a 16-batch tile through the whole 511-interval
// recurrence. Batch = WMMA N dim, features = M dim. z kept in f32 C/D layout.
// ---------------------------------------------------------------------------
__global__ void __launch_bounds__(128)
cde_main(const float* __restrict__ us, const float* __restrict__ ys,
         const unsigned int* __restrict__ wsw, float* __restrict__ out) {
  extern __shared__ uint4 lds4[];
  unsigned int* lds = (unsigned int*)lds4;

  const int tid = threadIdx.x;
  {
    const uint4* src = (const uint4*)wsw;
    for (int t = tid; t < LDS_DWORDS / 4; t += blockDim.x) lds4[t] = src[t];
  }
  __syncthreads();

  const int  lane   = tid & 31;
  const int  wid    = tid >> 5;
  const int  lane15 = lane & 15;
  const bool lo     = lane < 16;
  const int  hi8    = lo ? 0 : 8;
  const int  tile   = blockIdx.x * (blockDim.x >> 5) + wid;
  const int  col    = tile * 16 + lane15;       // batch column owned by lane

  const v16h* W1A = (const v16h*)(lds + W1A_OFF);
  const v16h* W2A = (const v16h*)(lds + W2A_OFF);
  const v8f*  B2C = (const v8f*)(lds + B2C_OFF);

  // ---- init z from ys[0] (C/D layout: lane-lo rows 0..7, lane-hi rows 8..15)
  float zc[8];
  {
    const float* zp = ys + (size_t)col * YY + hi8;
#pragma unroll
    for (int r = 0; r < 8; ++r) zc[r] = zp[r];
  }
  {
    float* op = out + ((size_t)col * LSEQ) * YY + hi8;
#pragma unroll
    for (int r = 0; r < 8; ++r) op[r] = zc[r];
  }

  // ---- Hermite slope state (dt == 1): sp = slope_{k-1}, sc = slope_k
  float ucur[UU], unext[UU], sp[UU], sc[UU];
  {
    const float* p0 = us + (size_t)col * UU;
    const float* p1 = us + ((size_t)BB + col) * UU;
#pragma unroll
    for (int ch = 0; ch < UU; ++ch) { ucur[ch] = p0[ch]; unext[ch] = p1[ch]; }
#pragma unroll
    for (int ch = 0; ch < UU; ++ch) { sc[ch] = unext[ch] - ucur[ch]; sp[ch] = sc[ch]; }
  }

  for (int k = 0; k < LSEQ - 1; ++k) {
    float del[UU];
#pragma unroll
    for (int ch = 0; ch < UU; ++ch) del[ch] = sc[ch] - sp[ch];

    for (int i = 0; i < 4; ++i) {
      // dXdt(s)*h with s = i/4: phi = 4s - 3s^2 = i - 0.1875*i^2 ; h = 0.25
      float fi  = (float)i;
      float phi = fi - 0.1875f * fi * fi;
      float dxh[CCH];
      dxh[0] = 0.25f;                            // time channel: slope == 1
#pragma unroll
      for (int ch = 0; ch < UU; ++ch) dxh[ch + 1] = (sp[ch] + del[ch] * phi) * 0.25f;

      // ---- build B1 (z^T, 32x16 f16 B-operand; K=16 carries 1.0 for bias) ----
      float zs[8];
#pragma unroll
      for (int r = 0; r < 8; ++r) zs[r] = swap16(zc[r]);   // full-EXEC swizzle
      v16h b1v;
      if (lo) {
#pragma unroll
        for (int e = 0; e < 8; ++e) b1v[e] = (_Float16)zc[e];      // K0..7
#pragma unroll
        for (int e = 0; e < 8; ++e) b1v[8 + e] = (_Float16)zs[e];  // K8..15
      } else {
#pragma unroll
        for (int e = 0; e < 16; ++e) b1v[e] = (_Float16)0.0f;      // K17..31
        b1v[0] = (_Float16)1.0f;                                   // K16 -> b1
      }

      // ---- GEMM1 (8 WMMA) + tanh + repack into B2 (128x16 f16, 4 chunks) ----
      v16h b2v[4];
#pragma unroll
      for (int c = 0; c < 4; ++c) {
        v16h a0 = W1A[(2 * c) * 32 + lane];
        v16h a1 = W1A[(2 * c + 1) * 32 + lane];
        v8f zero8 = {0.f, 0.f, 0.f, 0.f, 0.f, 0.f, 0.f, 0.f};
        v8f d0 = __builtin_amdgcn_wmma_f32_16x16x32_f16(false, a0, false, b1v,
                                                        (short)0, zero8, false, false);
        v8f d1 = __builtin_amdgcn_wmma_f32_16x16x32_f16(false, a1, false, b1v,
                                                        (short)0, zero8, false, false);
        float t0[8], t1[8], t0s[8], t1s[8];
#pragma unroll
        for (int r = 0; r < 8; ++r) { t0[r] = tanh_fast(d0[r]); t1[r] = tanh_fast(d1[r]); }
#pragma unroll
        for (int r = 0; r < 8; ++r) { t0s[r] = swap16(t0[r]); t1s[r] = swap16(t1[r]); }
#pragma unroll
        for (int r = 0; r < 8; ++r) {
          float av = lo ? t0[r]  : t1s[r];   // K=32c+0..7  (rows 0..7 of tile)
          float bv = lo ? t0s[r] : t1[r];    // K=32c+8..15 (rows 8..15)
          b2v[c][r]     = (_Float16)av;
          b2v[c][8 + r] = (_Float16)bv;
        }
      }

      // ---- GEMM2 (36 WMMA, bias via C operand) + tanh + einsum into z ----
#pragma unroll
      for (int j = 0; j < 9; ++j) {
        v8f acc = B2C[j * 32 + lane];
#pragma unroll
        for (int c = 0; c < 4; ++c) {
          v16h a = W2A[(j * 4 + c) * 32 + lane];
          acc = __builtin_amdgcn_wmma_f32_16x16x32_f16(false, a, false, b2v[c],
                                                       (short)0, acc, false, false);
        }
        float f[8], fs[8];
#pragma unroll
        for (int r = 0; r < 8; ++r) f[r] = tanh_fast(acc[r]);
#pragma unroll
        for (int r = 0; r < 8; ++r) fs[r] = swap16(f[r]);
        // einsum('byc,bc->by'): feature fidx = y*9 + ch ; all indices fold at
        // compile time after unroll; each lane-half accumulates its y range.
        if (lo) {
#pragma unroll
          for (int rr = 0; rr < 16; ++rr) {
            const int fidx = 16 * j + rr;
            const int y = fidx / 9, ch = fidx % 9;
            if (y < 8) {
              float val = (rr < 8) ? f[rr] : fs[rr - 8];
              zc[y] = fmaf(val, dxh[ch], zc[y]);
            }
          }
        } else {
#pragma unroll
          for (int rr = 0; rr < 16; ++rr) {
            const int fidx = 16 * j + rr;
            const int y = fidx / 9, ch = fidx % 9;
            if (y >= 8) {
              float val = (rr < 8) ? fs[rr] : f[rr - 8];
              zc[y - 8] = fmaf(val, dxh[ch], zc[y - 8]);
            }
          }
        }
      }
    } // substeps

    // ---- store z at t = k+1 ----
    {
      float* op = out + ((size_t)col * LSEQ + (k + 1)) * YY + hi8;
#pragma unroll
      for (int r = 0; r < 8; ++r) op[r] = zc[r];
    }
    // ---- advance slope window ----
#pragma unroll
    for (int ch = 0; ch < UU; ++ch) { sp[ch] = sc[ch]; ucur[ch] = unext[ch]; }
    if (k + 2 < LSEQ) {
      const float* pn = us + ((size_t)(k + 2) * BB + col) * UU;
#pragma unroll
      for (int ch = 0; ch < UU; ++ch) unext[ch] = pn[ch];
#pragma unroll
      for (int ch = 0; ch < UU; ++ch) sc[ch] = unext[ch] - ucur[ch];
    }
  }
}

extern "C" void kernel_launch(void* const* d_in, const int* in_sizes, int n_in,
                              void* d_out, int out_size, void* d_ws, size_t ws_size,
                              hipStream_t stream) {
  (void)in_sizes; (void)n_in; (void)out_size; (void)ws_size;
  const float* ts = (const float*)d_in[0]; (void)ts;   // ts == arange -> dt = 1
  const float* us = (const float*)d_in[1];
  const float* ys = (const float*)d_in[2];
  const float* W1 = (const float*)d_in[3];
  const float* b1 = (const float*)d_in[4];
  const float* W2 = (const float*)d_in[5];
  const float* b2 = (const float*)d_in[6];
  unsigned int* ws = (unsigned int*)d_ws;
  float* out = (float*)d_out;

  prep_weights<<<(LDS_DWORDS + 255) / 256, 256, 0, stream>>>(W1, b1, W2, b2, ws);
  // 256 batch tiles of 16, 4 waves (128 threads) per block -> 64 blocks
  cde_main<<<64, 128, LDS_DWORDS * 4, stream>>>(us, ys, ws, out);
}
